// EncoderAttention_39384850104482
// MI455X (gfx1250) — compile-verified, matrix operand order checked by reading, not run on hardware
//
#include <hip/hip_runtime.h>
#include <hip/hip_bf16.h>

// Problem constants (B=2, S=4096, D=1024 from the reference).
#define BB 2
#define SS 4096
#define DD 1024
#define TD 3072          // 3*D
#define KCH 64           // keys processed per online-softmax chunk
#define ATT_WAVES 2      // waves (q-tiles) per attention block

typedef __attribute__((ext_vector_type(8)))  __bf16 v8bf;
typedef __attribute__((ext_vector_type(16))) __bf16 v16bf;
typedef __attribute__((ext_vector_type(8)))  float  v8f;

static __device__ __forceinline__ v8f vzero8() {
  v8f z;
#pragma unroll
  for (int i = 0; i < 8; ++i) z[i] = 0.0f;
  return z;
}

// Load one 16x32 bf16 WMMA fragment (A layout; also used for B since we keep
// both operands "K-contiguous per lane"). Per the CDNA5 ISA 16-bit A layout:
// lane L holds row m = L%16; with h = L/16 it holds K in [8h,8h+8) and
// [16+8h,16+8h+8)  -> two contiguous 16-byte loads per lane.
static __device__ __forceinline__ v16bf ldfrag(const __bf16* base, int ld,
                                               int col0, int n, int h) {
  const __bf16* p = base + (size_t)n * ld + col0 + 8 * h;
  v8bf lo = *(const v8bf*)(p);
  v8bf hi = *(const v8bf*)(p + 16);
  v16bf r;
#pragma unroll
  for (int i = 0; i < 8; ++i) { r[i] = lo[i]; r[i + 8] = hi[i]; }
  return r;
}

static __device__ __forceinline__ v8f bf16_wmma(v16bf a, v16bf b, v8f c) {
  return __builtin_amdgcn_wmma_f32_16x16x32_bf16(
      /*neg_a=*/false, a, /*neg_b=*/false, b,
      /*c_mod=*/(short)0, c, /*reuse_a=*/false, /*reuse_b=*/false);
}

// ---------------------------------------------------------------------------
// Prep kernels: fp32 -> bf16 conversion (x) and transposed-weight conversion.
// ---------------------------------------------------------------------------
__global__ void cvt_x_kernel(const float* __restrict__ x,
                             __bf16* __restrict__ xb, int nelem) {
  int i = blockIdx.x * blockDim.x + threadIdx.x;
  if (i < nelem) xb[i] = (__bf16)x[i];
}

// Wt[c][d] = W[d][c]  (row-major [3D, D] bf16) so B-fragments are contiguous.
__global__ void cvt_wt_kernel(const float* __restrict__ W,
                              __bf16* __restrict__ Wt, int nelem) {
  int i = blockIdx.x * blockDim.x + threadIdx.x;
  if (i >= nelem) return;
  int c = i / DD;
  int d = i - c * DD;
  Wt[i] = (__bf16)W[(size_t)d * TD + c];
}

// ---------------------------------------------------------------------------
// QKV GEMM: [B*S, D] x [D, 3D] + bias. Each wave computes a 16x64 tile with
// 4 accumulators. Inner loop is software-pipelined (double-buffered fragment
// loads) so WMMAs overlap in-flight global loads instead of draining loadcnt.
// Writes Q,K row-major bf16 and V transposed bf16 [B, D, S].
// ---------------------------------------------------------------------------
__global__ void qkv_gemm_kernel(const __bf16* __restrict__ Xb,
                                const __bf16* __restrict__ Wt,
                                const float*  __restrict__ bias,
                                __bf16* __restrict__ Qb,
                                __bf16* __restrict__ Kb,
                                __bf16* __restrict__ Vt) {
  const int lane = threadIdx.x & 31;
  const int wave = threadIdx.x >> 5;
  const int n = lane & 15, h = lane >> 4;
  const int row0 = blockIdx.y * 16;                 // token tile
  const int col0 = blockIdx.x * 256 + wave * 64;    // output-column tile

  v8f acc[4];
#pragma unroll
  for (int j = 0; j < 4; ++j) acc[j] = vzero8();

  const __bf16* A = Xb + (size_t)row0 * DD;

  // Prologue: load fragments for d0 = 0.
  v16bf a_cur = ldfrag(A, DD, 0, n, h);
  v16bf b_cur[4];
#pragma unroll
  for (int j = 0; j < 4; ++j)
    b_cur[j] = ldfrag(Wt + (size_t)(col0 + 16 * j) * DD, DD, 0, n, h);

#pragma unroll 2
  for (int d0 = 0; d0 < DD - 32; d0 += 32) {
    // Preload next iteration's fragments before consuming current ones.
    v16bf a_nxt = ldfrag(A, DD, d0 + 32, n, h);
    v16bf b_nxt[4];
#pragma unroll
    for (int j = 0; j < 4; ++j)
      b_nxt[j] = ldfrag(Wt + (size_t)(col0 + 16 * j) * DD, DD, d0 + 32, n, h);
#pragma unroll
    for (int j = 0; j < 4; ++j) acc[j] = bf16_wmma(a_cur, b_cur[j], acc[j]);
    a_cur = a_nxt;
#pragma unroll
    for (int j = 0; j < 4; ++j) b_cur[j] = b_nxt[j];
  }
#pragma unroll
  for (int j = 0; j < 4; ++j) acc[j] = bf16_wmma(a_cur, b_cur[j], acc[j]);

#pragma unroll
  for (int j = 0; j < 4; ++j) {
    const int col = col0 + 16 * j + n;
    const int reg = (col0 + 16 * j) >> 10;          // uniform per tile: 0/1/2
    const float bv = bias[col];
#pragma unroll
    for (int r = 0; r < 8; ++r) {
      const int row = row0 + r + 8 * h;             // global token index
      const __bf16 v = (__bf16)(acc[j][r] + bv);
      if (reg == 0) {
        Qb[(size_t)row * DD + col] = v;
      } else if (reg == 1) {
        Kb[(size_t)row * DD + (col - DD)] = v;
      } else {
        const int bb = row / SS, s = row - bb * SS; // scatter to V^T
        Vt[((size_t)bb * DD + (col - 2 * DD)) * SS + s] = v;
      }
    }
  }
}

// ---------------------------------------------------------------------------
// Flash attention: one wave owns 16 queries x full D=1024. fp32 O accumulator
// lives in LDS (64KB/wave); online softmax rescale fused into the C-read of
// the P*V WMMA. P is re-laid-out C->A via a 2KB LDS staging tile. Both WMMA
// loops are double-buffered for load/compute overlap.
// ---------------------------------------------------------------------------
__global__ void attn_kernel(const __bf16* __restrict__ Qb,
                            const __bf16* __restrict__ Kb,
                            const __bf16* __restrict__ Vt,
                            float* __restrict__ out) {
  extern __shared__ char smem_raw[];
  const int lane = threadIdx.x & 31;
  const int wave = threadIdx.x >> 5;
  const int n = lane & 15, h = lane >> 4;

  float*  Osh = (float*)smem_raw + (size_t)wave * 16 * DD;
  __bf16* Pst = (__bf16*)(smem_raw + (size_t)ATT_WAVES * 16 * DD * sizeof(float))
                + (size_t)wave * 16 * KCH;

  const int tile = blockIdx.x * ATT_WAVES + wave;
  const int b  = tile / (SS / 16);
  const int q0 = (tile % (SS / 16)) * 16;

  const __bf16* Q = Qb + ((size_t)b * SS + q0) * DD;  // 16 x D
  const __bf16* K = Kb + (size_t)b * SS * DD;         // S x D
  const __bf16* V = Vt + (size_t)b * DD * SS;         // D x S (transposed)

  for (int i = lane; i < 16 * DD; i += 32) Osh[i] = 0.0f;

  const float scale = 0.03125f;  // 1/sqrt(1024)
  float mrow[8], lrow[8];
#pragma unroll
  for (int r = 0; r < 8; ++r) { mrow[r] = -3.0e38f; lrow[r] = 0.0f; }

  for (int kc = 0; kc < SS; kc += KCH) {
    // ---- scores: 4 tiles of 16x16, contracting over D, double-buffered ----
    v8f acc[4];
#pragma unroll
    for (int j = 0; j < 4; ++j) acc[j] = vzero8();

    v16bf a_cur = ldfrag(Q, DD, 0, n, h);
    v16bf bk_cur[4];
#pragma unroll
    for (int j = 0; j < 4; ++j)
      bk_cur[j] = ldfrag(K + (size_t)(kc + 16 * j) * DD, DD, 0, n, h);

#pragma unroll 2
    for (int d0 = 0; d0 < DD - 32; d0 += 32) {
      v16bf a_nxt = ldfrag(Q, DD, d0 + 32, n, h);
      v16bf bk_nxt[4];
#pragma unroll
      for (int j = 0; j < 4; ++j)
        bk_nxt[j] = ldfrag(K + (size_t)(kc + 16 * j) * DD, DD, d0 + 32, n, h);
#pragma unroll
      for (int j = 0; j < 4; ++j) acc[j] = bf16_wmma(a_cur, bk_cur[j], acc[j]);
      a_cur = a_nxt;
#pragma unroll
      for (int j = 0; j < 4; ++j) bk_cur[j] = bk_nxt[j];
    }
#pragma unroll
    for (int j = 0; j < 4; ++j) acc[j] = bf16_wmma(a_cur, bk_cur[j], acc[j]);

    // ---- online softmax over this 64-key chunk (rows span 16 lanes) ----
    float alpha[8];
#pragma unroll
    for (int r = 0; r < 8; ++r) {
      float mx = -3.0e38f;
#pragma unroll
      for (int j = 0; j < 4; ++j) mx = fmaxf(mx, acc[j][r] * scale);
#pragma unroll
      for (int off = 1; off < 16; off <<= 1)
        mx = fmaxf(mx, __shfl_xor(mx, off, 32));
      const float mnew = fmaxf(mrow[r], mx);
      alpha[r] = __expf(mrow[r] - mnew);
      mrow[r] = mnew;
      float sum = 0.0f;
#pragma unroll
      for (int j = 0; j < 4; ++j) {
        const float p = __expf(acc[j][r] * scale - mnew);
        sum += p;
        // stage P row-major 16x64 so it can be reloaded in A-fragment layout
        Pst[(r + 8 * h) * KCH + 16 * j + n] = (__bf16)p;
      }
#pragma unroll
      for (int off = 1; off < 16; off <<= 1) sum += __shfl_xor(sum, off, 32);
      lrow[r] = lrow[r] * alpha[r] + sum;
    }

    // ---- P*V: A = P (16x64 from LDS), B = V^T rows, C = rescaled O ----
    v16bf pa0 = ldfrag(Pst, KCH, 0,  n, h);
    v16bf pa1 = ldfrag(Pst, KCH, 32, n, h);

    v16bf vb0c = ldfrag(V, SS, kc,      n, h);
    v16bf vb1c = ldfrag(V, SS, kc + 32, n, h);
#pragma unroll 2
    for (int dt = 0; dt < DD - 16; dt += 16) {
      v16bf vb0n = ldfrag(V + (size_t)(dt + 16) * SS, SS, kc,      n, h);
      v16bf vb1n = ldfrag(V + (size_t)(dt + 16) * SS, SS, kc + 32, n, h);
      v8f c;
#pragma unroll
      for (int r = 0; r < 8; ++r)
        c[r] = Osh[(r + 8 * h) * DD + dt + n] * alpha[r];  // fused rescale
      c = bf16_wmma(pa0, vb0c, c);
      c = bf16_wmma(pa1, vb1c, c);
#pragma unroll
      for (int r = 0; r < 8; ++r)
        Osh[(r + 8 * h) * DD + dt + n] = c[r];
      vb0c = vb0n;
      vb1c = vb1n;
    }
    {
      const int dt = DD - 16;
      v8f c;
#pragma unroll
      for (int r = 0; r < 8; ++r)
        c[r] = Osh[(r + 8 * h) * DD + dt + n] * alpha[r];
      c = bf16_wmma(pa0, vb0c, c);
      c = bf16_wmma(pa1, vb1c, c);
#pragma unroll
      for (int r = 0; r < 8; ++r)
        Osh[(r + 8 * h) * DD + dt + n] = c[r];
    }
  }

  // ---- epilogue: normalize by row sum, write fp32 output ----
  float* op = out + ((size_t)b * SS + q0) * DD;
#pragma unroll
  for (int r = 0; r < 8; ++r) {
    const float inv = 1.0f / lrow[r];
    const int row = r + 8 * h;
    for (int c0 = n; c0 < DD; c0 += 16)
      op[(size_t)row * DD + c0] = Osh[row * DD + c0] * inv;
  }
}

// ---------------------------------------------------------------------------
extern "C" void kernel_launch(void* const* d_in, const int* in_sizes, int n_in,
                              void* d_out, int out_size, void* d_ws, size_t ws_size,
                              hipStream_t stream) {
  const float* x    = (const float*)d_in[0];   // [B, S, D]
  const float* W    = (const float*)d_in[1];   // [D, 3D]
  const float* bias = (const float*)d_in[2];   // [3D]
  float* out = (float*)d_out;                  // [B, S, D]

  char* ws = (char*)d_ws;
  size_t off = 0;
  __bf16* Xb = (__bf16*)(ws + off); off += (size_t)BB * SS * DD * 2;  // 16 MB
  __bf16* Wt = (__bf16*)(ws + off); off += (size_t)TD * DD * 2;       //  6 MB
  __bf16* Qb = (__bf16*)(ws + off); off += (size_t)BB * SS * DD * 2;  // 16 MB
  __bf16* Kb = (__bf16*)(ws + off); off += (size_t)BB * SS * DD * 2;  // 16 MB
  __bf16* Vt = (__bf16*)(ws + off); off += (size_t)BB * SS * DD * 2;  // 16 MB

  const int nx = BB * SS * DD;
  cvt_x_kernel<<<(nx + 255) / 256, 256, 0, stream>>>(x, Xb, nx);
  const int nw = TD * DD;
  cvt_wt_kernel<<<(nw + 255) / 256, 256, 0, stream>>>(W, Wt, nw);

  // 16-row x 256-col tile per 4-wave block over the [B*S, 3D] output.
  qkv_gemm_kernel<<<dim3(TD / 256, BB * SS / 16), 128, 0, stream>>>(
      Xb, Wt, bias, Qb, Kb, Vt);

  const size_t lds = (size_t)ATT_WAVES *
                     (16 * DD * sizeof(float) + 16 * KCH * sizeof(__bf16)); // 132 KB
  attn_kernel<<<(BB * SS / 16) / ATT_WAVES, ATT_WAVES * 32, lds, stream>>>(
      Qb, Kb, Vt, out);
}
